// ASTGCNHead_86655260164728
// MI455X (gfx1250) — matrix operand order repaired and checked by reading, not compiled, not run on hardware
//
#include <hip/hip_runtime.h>
#include <cstdint>
#include <cstddef>

// ---------------------------------------------------------------------------
// ASTGCN forward for MI455X (gfx1250, wave32, WMMA).
// Identity adjacency -> cheb = [I,-I,I] (diagonal): graph conv needs only
// diag(softmax(S)) and ThetaEff = T0-T1+T2.
// Heavy math on v_wmma_f32_16x16x32_bf16 (f32 accumulate):
//   - per-batch 1000^3 GEMM (sVs @ sigmoid(...)), softmax stats fused into the
//     WMMA epilogue (exp column-sums via atomics + diagonal), raw S never
//     materialized.
//   - 48000 x F x 64 Theta projection with fused diag(S)-scale + relu.
// Both WMMA kernels use 16x64 register blocking: A fragment loaded once per
// k-step, reused across 4 B fragments (10 loads / 4 WMMAs).
// ---------------------------------------------------------------------------

typedef __bf16 bf16_t;
typedef __bf16 bf16x8  __attribute__((ext_vector_type(8)));
typedef __bf16 bf16x16 __attribute__((ext_vector_type(16)));
typedef float  f32x8   __attribute__((ext_vector_type(8)));

constexpr int B  = 4;
constexpr int N  = 1000;
constexpr int T  = 12;
constexpr int D  = 128;   // block0 feature dim
constexpr int CC = 64;    // block1 feature dim / channel count
constexpr int P  = 12;
constexpr int NP = 1024;  // N padded to WMMA tiles
constexpr int MROWS = B * N * T; // 48000 (divisible by 16)

__device__ __forceinline__ bf16x16 combine16(bf16x8 lo, bf16x8 hi) {
  return __builtin_shufflevector(lo, hi, 0,1,2,3,4,5,6,7,8,9,10,11,12,13,14,15);
}

// ---- layout pack: x (B,N,T,D) -> X0 (B,N,D,T) --------------------------------
__global__ void k_pack_x0(const float* __restrict__ xin, float* __restrict__ X0) {
  int idx = blockIdx.x * 256 + threadIdx.x;               // B*N*D*T, exact grid
  int t = idx % T;
  int f = (idx / T) % D;
  int bn = idx / (D * T);
  X0[idx] = xin[((size_t)bn * T + t) * D + f];
}

// ---- rhs[b,n,t] = sum_f tU3[f] * X[b,n,f,t] ---------------------------------
__global__ void k_rhs(const float* __restrict__ X, const float* __restrict__ tU3,
                      float* __restrict__ rhs, int F, int total) {
  int i = blockIdx.x * blockDim.x + threadIdx.x;
  if (i >= total) return;
  int t = i % T;
  int bn = i / T;
  const float* xp = X + ((size_t)bn * F) * T + t;
  float a = 0.f;
  for (int f = 0; f < F; ++f) a += tU3[f] * xp[(size_t)f * T];
  rhs[i] = a;
}

// ---- xu[b,f,t] = sum_n X[b,n,f,t] * tU1[n] ----------------------------------
__global__ void k_xu(const float* __restrict__ X, const float* __restrict__ tU1,
                     float* __restrict__ xu, int F) {
  int i = blockIdx.x * 256 + threadIdx.x;                 // B*F*T, exact grid
  int t = i % T;
  int f = (i / T) % F;
  int b = i / (F * T);
  const float* xp = X + (((size_t)b * N) * F + f) * T + t;
  float a = 0.f;
  for (int n = 0; n < N; ++n) a += xp[(size_t)n * F * T] * tU1[n];
  xu[i] = a;
}

// ---- M[b,f,s] = sum_n tU2[f,n] * rhs[b,n,s] ---------------------------------
__global__ void k_M(const float* __restrict__ rhs, const float* __restrict__ tU2,
                    float* __restrict__ Mb, int F) {
  int i = blockIdx.x * 256 + threadIdx.x;                 // B*F*T, exact grid
  int s = i % T;
  int f = (i / T) % F;
  int b = i / (F * T);
  const float* rp = rhs + (size_t)b * N * T + s;
  const float* up = tU2 + (size_t)f * N;
  float a = 0.f;
  for (int n = 0; n < N; ++n) a += up[n] * rp[(size_t)n * T];
  Mb[i] = a;
}

// ---- temporal attention E (per batch, tiny) ---------------------------------
__global__ void k_E(const float* __restrict__ xu, const float* __restrict__ Mb,
                    const float* __restrict__ tbe, const float* __restrict__ tVe,
                    float* __restrict__ E, int F) {
  int b = blockIdx.x;
  int tid = threadIdx.x;                                  // 144 threads
  int t = tid / T, s = tid % T;
  __shared__ float sig[T * T];
  __shared__ float e0[T * T];
  const float* xub = xu + (size_t)b * F * T;
  const float* Mbb = Mb + (size_t)b * F * T;
  float p = 0.f;
  for (int f = 0; f < F; ++f) p += xub[f * T + t] * Mbb[f * T + s];
  p += tbe[t * T + s];
  sig[tid] = 1.f / (1.f + __expf(-p));
  __syncthreads();
  float acc = 0.f;
  for (int u = 0; u < T; ++u) acc += tVe[t * T + u] * sig[u * T + s];
  e0[tid] = __expf(acc);
  __syncthreads();
  float den = 0.f;
  for (int u = 0; u < T; ++u) den += e0[u * T + s];
  E[b * T * T + tid] = e0[tid] / den;                     // softmax over axis t
}

// ---- x_t[b,n,f,s] = sum_t X[b,n,f,t]*E[b,t,s]; also xtsum = x_t . sW1 -------
__global__ void k_xt(const float* __restrict__ X, const float* __restrict__ E,
                     const float* __restrict__ sW1, float* __restrict__ XT,
                     float* __restrict__ xtsum, int F) {
  int idx = blockIdx.x * 256 + threadIdx.x;               // B*N*F, exact grid
  int b = idx / (N * F);                                  // constant per block
  __shared__ float Es[T * T];
  if (threadIdx.x < T * T) Es[threadIdx.x] = E[b * T * T + threadIdx.x];
  __syncthreads();
  const float* xr = X + (size_t)idx * T;
  float xv[T];
#pragma unroll
  for (int t = 0; t < T; ++t) xv[t] = xr[t];
  float* o = XT + (size_t)idx * T;
  float ss = 0.f;
#pragma unroll
  for (int s = 0; s < T; ++s) {
    float a = 0.f;
#pragma unroll
    for (int t = 0; t < T; ++t) a += xv[t] * Es[t * T + s];
    o[s] = a;
    ss += a * sW1[s];
  }
  xtsum[idx] = ss;
}

// ---- slhs / srhs ------------------------------------------------------------
__global__ void k_ssr(const float* __restrict__ XT, const float* __restrict__ xts,
                      const float* __restrict__ sW2, const float* __restrict__ sW3,
                      float* __restrict__ slhs, float* __restrict__ srhs,
                      int F, int total) {
  int i = blockIdx.x * blockDim.x + threadIdx.x;
  if (i >= total) return;
  int t = i % T;
  int bn = i / T;
  const float* xs = xts + (size_t)bn * F;
  const float* xp = XT + ((size_t)bn * F) * T + t;
  float a = 0.f, r = 0.f;
  for (int f = 0; f < F; ++f) {
    a += xs[f] * sW2[f * T + t];
    r += sW3[f] * xp[(size_t)f * T];
  }
  slhs[i] = a;
  srhs[i] = r;
}

// ---- sigT[b][k][m] = sigmoid(<slhs_m, srhs_k> + sbs[m,k]), bf16, padded -----
__global__ void k_sig(const float* __restrict__ slhs, const float* __restrict__ srhs,
                      const float* __restrict__ sbs, bf16_t* __restrict__ sigT) {
  int b = blockIdx.y;
  int idx = blockIdx.x * 256 + threadIdx.x;               // NP*NP, exact grid
  int k = idx >> 10;
  int m = idx & (NP - 1);
  float v = 0.f;
  if (m < N && k < N) {
    const float* lm = slhs + ((size_t)b * N + m) * T;
    const float* rk = srhs + ((size_t)b * N + k) * T;
    float a = sbs[(size_t)m * N + k];
#pragma unroll
    for (int t = 0; t < T; ++t) a += lm[t] * rk[t];
    v = 1.f / (1.f + __expf(-a));
  }
  sigT[(size_t)b * NP * NP + (size_t)k * NP + m] = (bf16_t)v;
}

// ---- sVs -> padded bf16 -----------------------------------------------------
__global__ void k_cvt_sVs(const float* __restrict__ sVs, bf16_t* __restrict__ out) {
  int idx = blockIdx.x * 256 + threadIdx.x;               // NP*NP, exact grid
  int n = idx >> 10;
  int m = idx & (NP - 1);
  float v = (n < N && m < N) ? sVs[(size_t)n * N + m] : 0.f;
  out[idx] = (bf16_t)v;
}

__global__ void k_zero_cd(float* __restrict__ colsum, float* __restrict__ diag) {
  int i = blockIdx.x * 256 + threadIdx.x;
  if (i < B * NP) { colsum[i] = 0.f; diag[i] = 0.f; }
}

// ---- WMMA GEMM: Sraw = sVs(bf16) @ sigT^T per batch; fused exp/colsum/diag --
// Register-blocked: each wave computes a 16x64 strip (1 A fragment reused
// across 4 B fragments per k-step).
__global__ void k_wmma_S(const bf16_t* __restrict__ A, const bf16_t* __restrict__ Bm,
                         float* __restrict__ colsum, float* __restrict__ diag) {
  int wave = threadIdx.x >> 5;
  int lane = threadIdx.x & 31;
  int task = blockIdx.x * 8 + wave;                       // B*64*16 strip tasks
  int b   = task >> 10;                                   // 1024 strips / batch
  int t2  = task & 1023;
  int tm  = t2 >> 4;                                      // 64 row tiles
  int tn0 = (t2 & 15) * 4;                                // first of 4 col tiles
  int half = lane >> 4, l15 = lane & 15;
  const bf16_t* arow = A + (size_t)(tm * 16 + l15) * NP;
  const bf16_t* brow0 = Bm + (size_t)b * NP * NP + (size_t)(tn0 * 16 + l15) * NP;
  f32x8 acc0 = {0.f,0.f,0.f,0.f,0.f,0.f,0.f,0.f};
  f32x8 acc1 = acc0, acc2 = acc0, acc3 = acc0;
  for (int kk = 0; kk < NP; kk += 32) {
    int ao = kk + half * 16;                              // A: K = half*16+0..15
    int bo = kk + half * 16;                              // B: K = half*16+0..15
    __builtin_prefetch((const void*)(arow + ao + 256), 0, 1);
    bf16x8 a0 = *(const bf16x8*)(arow + ao);
    bf16x8 a1 = *(const bf16x8*)(arow + ao + 8);
    bf16x16 av = combine16(a0, a1);
#pragma unroll
    for (int j = 0; j < 4; ++j) {
      const bf16_t* br = brow0 + (size_t)j * 16 * NP;
      __builtin_prefetch((const void*)(br + bo + 256), 0, 1);
      bf16x8 b0 = *(const bf16x8*)(br + bo);
      bf16x8 b1 = *(const bf16x8*)(br + bo + 8);
      bf16x16 bv = combine16(b0, b1);
      f32x8 c = (j == 0) ? acc0 : (j == 1) ? acc1 : (j == 2) ? acc2 : acc3;
      c = __builtin_amdgcn_wmma_f32_16x16x32_bf16(false, av, false, bv,
                                                  (short)0, c, false, false);
      if (j == 0) acc0 = c; else if (j == 1) acc1 = c;
      else if (j == 2) acc2 = c; else acc3 = c;
    }
  }
#pragma unroll
  for (int j = 0; j < 4; ++j) {
    f32x8 c = (j == 0) ? acc0 : (j == 1) ? acc1 : (j == 2) ? acc2 : acc3;
    int col = (tn0 + j) * 16 + l15;
    float lsum = 0.f;
#pragma unroll
    for (int r = 0; r < 8; ++r) {
      int row = tm * 16 + half * 8 + r;
      float e = __expf(c[r]);
      if (row < N && col < N) {
        lsum += e;
        if (row == col) diag[b * NP + row] = e;
      }
    }
    if (col < N) atomicAdd(&colsum[b * NP + col], lsum);
  }
}

// ---- ThetaEff^T (CC x F) bf16 ----------------------------------------------
__global__ void k_theta_eff(const float* __restrict__ Th, bf16_t* __restrict__ ThT,
                            int F) {
  int idx = blockIdx.x * 256 + threadIdx.x;
  if (idx >= F * CC) return;
  int f = idx / CC, o = idx % CC;
  float v = Th[(size_t)f * CC + o] - Th[(size_t)F * CC + (size_t)f * CC + o]
          + Th[2 * (size_t)F * CC + (size_t)f * CC + o];
  ThT[(size_t)o * F + f] = (bf16_t)v;
}

// ---- X -> Xb bf16, rows r=(b*N+m)*T+t, cols f -------------------------------
__global__ void k_pack_Xb(const float* __restrict__ X, bf16_t* __restrict__ Xb,
                          int F) {
  int idx = blockIdx.x * 256 + threadIdx.x;               // MROWS*F, exact grid
  int f = idx % F;
  int r = idx / F;
  int t = r % T;
  int bm = r / T;
  Xb[idx] = (bf16_t)X[((size_t)bm * F + f) * T + t];
}

// ---- WMMA GEMM: D = Xb @ ThetaEff; fused diag(S) scale + relu -> G[b,c,n,t] -
// Each wave computes one full 16x64 output strip (all 4 col tiles).
__global__ void k_wmma_G(const bf16_t* __restrict__ Xb, const bf16_t* __restrict__ ThT,
                         const float* __restrict__ colsum, const float* __restrict__ diag,
                         float* __restrict__ G, int F) {
  int wave = threadIdx.x >> 5;
  int lane = threadIdx.x & 31;
  int tr = blockIdx.x * 8 + wave;                         // MROWS/16 strip tasks
  int half = lane >> 4, l15 = lane & 15;
  const bf16_t* arow = Xb + (size_t)(tr * 16 + l15) * F;
  const bf16_t* brow0 = ThT + (size_t)l15 * F;
  f32x8 acc0 = {0.f,0.f,0.f,0.f,0.f,0.f,0.f,0.f};
  f32x8 acc1 = acc0, acc2 = acc0, acc3 = acc0;
  for (int kk = 0; kk < F; kk += 32) {
    int ao = kk + half * 16;
    int bo = kk + half * 16;
    bf16x8 a0 = *(const bf16x8*)(arow + ao);
    bf16x8 a1 = *(const bf16x8*)(arow + ao + 8);
    bf16x16 av = combine16(a0, a1);
#pragma unroll
    for (int j = 0; j < 4; ++j) {
      const bf16_t* br = brow0 + (size_t)j * 16 * F;
      bf16x8 b0 = *(const bf16x8*)(br + bo);
      bf16x8 b1 = *(const bf16x8*)(br + bo + 8);
      bf16x16 bv = combine16(b0, b1);
      f32x8 c = (j == 0) ? acc0 : (j == 1) ? acc1 : (j == 2) ? acc2 : acc3;
      c = __builtin_amdgcn_wmma_f32_16x16x32_bf16(false, av, false, bv,
                                                  (short)0, c, false, false);
      if (j == 0) acc0 = c; else if (j == 1) acc1 = c;
      else if (j == 2) acc2 = c; else acc3 = c;
    }
  }
#pragma unroll
  for (int j = 0; j < 4; ++j) {
    f32x8 c = (j == 0) ? acc0 : (j == 1) ? acc1 : (j == 2) ? acc2 : acc3;
    int o = j * 16 + l15;
#pragma unroll
    for (int r = 0; r < 8; ++r) {
      int gr = tr * 16 + half * 8 + r;
      int t = gr % T;
      int bm = gr / T;
      int b = bm / N, m = bm % N;
      float sd = diag[b * NP + m] / colsum[b * NP + m];
      float v = sd * c[r];
      G[(((size_t)b * CC + o) * N + m) * T + t] = v > 0.f ? v : 0.f;
    }
  }
}

// ---- time conv (1x3 on T) + 1x1 residual + relu + LayerNorm over CC ---------
__global__ void k_tconv_ln(const float* __restrict__ G, const float* __restrict__ X,
                           const float* __restrict__ tcw, const float* __restrict__ tcb,
                           const float* __restrict__ rcw, const float* __restrict__ rcb,
                           const float* __restrict__ lng, const float* __restrict__ lnb,
                           float* __restrict__ Xout, int F) {
  int bnt = blockIdx.x;                                   // B*N*T blocks, 64 thr
  int t = bnt % T;
  int bn = bnt / T;
  int n = bn % N;
  int b = bn / N;
  int o = threadIdx.x;                                    // channel 0..63
  __shared__ float sG[3 * CC];
  __shared__ float sX[D];
  __shared__ float red[CC];
#pragma unroll
  for (int j = 0; j < 3; ++j) {
    int tt = t + j - 1;
    sG[j * CC + o] = (tt >= 0 && tt < T)
        ? G[(((size_t)b * CC + o) * N + n) * T + tt] : 0.f;
  }
  for (int c = o; c < F; c += CC)
    sX[c] = X[(((size_t)b * N + n) * F + c) * T + t];
  __syncthreads();
  float tc = tcb[o];
  for (int c = 0; c < CC; ++c) {
    const float* wv = tcw + ((size_t)o * CC + c) * 3;
    tc += wv[0] * sG[c] + wv[1] * sG[CC + c] + wv[2] * sG[2 * CC + c];
  }
  float rs = rcb[o];
  for (int c = 0; c < F; ++c) rs += rcw[(size_t)o * F + c] * sX[c];
  float v = tc + rs;
  v = v > 0.f ? v : 0.f;
  red[o] = v;
  __syncthreads();
  for (int s = 32; s > 0; s >>= 1) { if (o < s) red[o] += red[o + s]; __syncthreads(); }
  float mu = red[0] * (1.f / CC);
  __syncthreads();
  float dv = v - mu;
  red[o] = dv * dv;
  __syncthreads();
  for (int s = 32; s > 0; s >>= 1) { if (o < s) red[o] += red[o + s]; __syncthreads(); }
  float var = red[0] * (1.f / CC);
  float y = dv * rsqrtf(var + 1e-5f) * lng[o] + lnb[o];
  Xout[(((size_t)b * N + n) * CC + o) * T + t] = y;
}

// ---- final FC: out[b,n,p] = fcb[p] + sum_{t,f} X2[b,n,f,t]*fcw[p,t,f] -------
__global__ void k_fc(const float* __restrict__ X2, const float* __restrict__ fcw,
                     const float* __restrict__ fcb, float* __restrict__ out) {
  int i = blockIdx.x * blockDim.x + threadIdx.x;
  if (i >= B * N * P) return;
  int p = i % P;
  int bn = i / P;
  const float* xp = X2 + (size_t)bn * CC * T;
  float a = fcb[p];
  for (int t = 0; t < T; ++t) {
    const float* wp = fcw + ((size_t)p * T + t) * CC;
    for (int f = 0; f < CC; ++f) a += xp[(size_t)f * T + t] * wp[f];
  }
  out[i] = a;
}

// ---------------------------------------------------------------------------
struct Scratch {
  float *rhs, *slhs, *srhs, *xu, *Mb, *E, *xts, *G, *colsum, *diag;
  bf16_t *sigT, *sVsb, *Xb, *ThT;
};

static void launch_block(const float* X, float* XT, float* Xout, int F,
                         void* const* d_in, int pb, const Scratch& S,
                         hipStream_t st) {
  const float* tU1 = (const float*)d_in[pb + 0];
  const float* tU2 = (const float*)d_in[pb + 1];
  const float* tU3 = (const float*)d_in[pb + 2];
  const float* tbe = (const float*)d_in[pb + 3];
  const float* tVe = (const float*)d_in[pb + 4];
  const float* sW1 = (const float*)d_in[pb + 5];
  const float* sW2 = (const float*)d_in[pb + 6];
  const float* sW3 = (const float*)d_in[pb + 7];
  const float* sbs = (const float*)d_in[pb + 8];
  const float* sVs = (const float*)d_in[pb + 9];
  const float* Th  = (const float*)d_in[pb + 10];
  const float* tcw = (const float*)d_in[pb + 11];
  const float* tcb = (const float*)d_in[pb + 12];
  const float* rcw = (const float*)d_in[pb + 13];
  const float* rcb = (const float*)d_in[pb + 14];
  const float* lng = (const float*)d_in[pb + 15];
  const float* lnb = (const float*)d_in[pb + 16];

  int tot = B * N * T;
  k_rhs<<<(tot + 255) / 256, 256, 0, st>>>(X, tU3, S.rhs, F, tot);
  k_xu<<<(B * F * T) / 256, 256, 0, st>>>(X, tU1, S.xu, F);
  k_M<<<(B * F * T) / 256, 256, 0, st>>>(S.rhs, tU2, S.Mb, F);
  k_E<<<B, T * T, 0, st>>>(S.xu, S.Mb, tbe, tVe, S.E, F);
  k_xt<<<(B * N * F) / 256, 256, 0, st>>>(X, S.E, sW1, XT, S.xts, F);
  k_ssr<<<(tot + 255) / 256, 256, 0, st>>>(XT, S.xts, sW2, sW3, S.slhs, S.srhs, F, tot);
  k_sig<<<dim3(NP * NP / 256, B), 256, 0, st>>>(S.slhs, S.srhs, sbs, S.sigT);
  k_cvt_sVs<<<NP * NP / 256, 256, 0, st>>>(sVs, S.sVsb);
  k_zero_cd<<<(B * NP + 255) / 256, 256, 0, st>>>(S.colsum, S.diag);
  k_wmma_S<<<B * 64 * 16 / 8, 256, 0, st>>>(S.sVsb, S.sigT, S.colsum, S.diag);
  k_theta_eff<<<(F * CC + 255) / 256, 256, 0, st>>>(Th, S.ThT, F);
  k_pack_Xb<<<(MROWS * F) / 256, 256, 0, st>>>(X, S.Xb, F);
  k_wmma_G<<<(MROWS / 16) / 8, 256, 0, st>>>(S.Xb, S.ThT, S.colsum, S.diag, S.G, F);
  k_tconv_ln<<<B * N * T, CC, 0, st>>>(S.G, X, tcw, tcb, rcw, rcb, lng, lnb, Xout, F);
}

extern "C" void kernel_launch(void* const* d_in, const int* in_sizes, int n_in,
                              void* d_out, int out_size, void* d_ws, size_t ws_size,
                              hipStream_t stream) {
  (void)in_sizes; (void)n_in; (void)out_size; (void)ws_size;
  char* w = (char*)d_ws;
  size_t off = 0;
  auto take = [&](size_t bytes) -> void* {
    void* p = w + off;
    off += (bytes + 255) & ~(size_t)255;
    return p;
  };
  // big ping-pong buffers
  float* bufA = (float*)take((size_t)B * N * D * T * 4);  // X0 / final block-1 out
  float* bufB = (float*)take((size_t)B * N * D * T * 4);  // x_t (both blocks)
  float* bufC = (float*)take((size_t)B * N * CC * T * 4); // block-0 out / block-1 in
  Scratch S;
  S.Xb     = (bf16_t*)take((size_t)MROWS * D * 2);
  S.sigT   = (bf16_t*)take((size_t)B * NP * NP * 2);
  S.sVsb   = (bf16_t*)take((size_t)NP * NP * 2);
  S.ThT    = (bf16_t*)take((size_t)CC * D * 2);
  S.G      = (float*)take((size_t)B * CC * N * T * 4);
  S.rhs    = (float*)take((size_t)B * N * T * 4);
  S.slhs   = (float*)take((size_t)B * N * T * 4);
  S.srhs   = (float*)take((size_t)B * N * T * 4);
  S.xu     = (float*)take((size_t)B * D * T * 4);
  S.Mb     = (float*)take((size_t)B * D * T * 4);
  S.E      = (float*)take((size_t)B * T * T * 4);
  S.xts    = (float*)take((size_t)B * N * D * 4);
  S.colsum = (float*)take((size_t)B * NP * 4);
  S.diag   = (float*)take((size_t)B * NP * 4);

  const float* xin = (const float*)d_in[0];
  k_pack_x0<<<(B * N * D * T) / 256, 256, 0, stream>>>(xin, bufA);

  launch_block(bufA, bufB, bufC, D,  d_in, 1,  S, stream);   // block 0 (F=128)
  launch_block(bufC, bufB, bufA, CC, d_in, 18, S, stream);   // block 1 (F=64)

  k_fc<<<(B * N * P + 255) / 256, 256, 0, stream>>>(
      bufA, (const float*)d_in[35], (const float*)d_in[36], (float*)d_out);
}
